// SchedulerNetwork_19971597926440
// MI455X (gfx1250) — matrix operand order repaired and checked by reading, not compiled, not run on hardware
//
#include <hip/hip_runtime.h>
#include <hip/hip_bf16.h>

// ---------------------------------------------------------------------------
// SchedulerNetwork for MI455X (gfx1250, wave32, WMMA).
//  - All 256/512-K GEMMs via v_wmma_f32_16x16x32_f16 (f16 A/B, f32 acc).
//  - Each wave computes a 32x32 C tile: 4 WMMAs per 4 fragment loads
//    (2x the arithmetic intensity of a 16x32 strip).
//  - B matrices stored transposed (N x K) so every lane fragment is a
//    contiguous 16B/32B global load.
//  - reward/idle MLP factorized: fin@W1 = A_r[i] + A_t[j] + pd*w_last + b1.
// ---------------------------------------------------------------------------

typedef _Float16 h16;
typedef __attribute__((ext_vector_type(16))) _Float16 v16h;
typedef __attribute__((ext_vector_type(8)))  _Float16 v8h;
typedef __attribute__((ext_vector_type(8)))  float    v8f;

#define E_   256
#define FF_  512
#define B_   8
#define N_   64
#define M_   128

// ------------------------------ WMMA GEMM ----------------------------------
// C[M,N](f32) = A[M,K](f16) @ B^T[N,K](f16) (+bias) (+C if acc) (+activation)
// Batched: batch = BO * BI with independent strides for A/B/C.
// Each wave computes a 32x32 tile of C (2 M-tiles x 2 N-tiles, 4 wmma/K-step).
__global__ __launch_bounds__(256)
void wmma_gemm(const h16* __restrict__ A, int lda, long sAo, long sAi,
               const h16* __restrict__ Bt, int ldb, long sBo, long sBi,
               float* __restrict__ C, int ldc, long sCo, long sCi,
               h16* __restrict__ C16,
               const float* __restrict__ bias,
               int Mr, int Nr, int Kr, int BO, int BI, int act, int accum)
{
    int wave = blockIdx.x * 8 + (threadIdx.x >> 5);
    int lane = threadIdx.x & 31;
    int tilesM2 = Mr >> 5;              // 32 rows per wave
    int tilesN2 = Nr >> 5;              // 32 cols per wave
    int per   = tilesM2 * tilesN2;
    int total = BO * BI * per;
    if (wave >= total) return;          // wave-uniform branch: EXEC stays ~0

    int bidx = wave / per;
    int t    = wave - bidx * per;
    int tm   = t / tilesN2;
    int tn   = t - tm * tilesN2;
    int bo   = bidx / BI;
    int bi   = bidx - bo * BI;

    const h16* Ab = A  + (long)bo * sAo + (long)bi * sAi;
    const h16* Bb = Bt + (long)bo * sBo + (long)bi * sBi;
    long cOff     = (long)bo * sCo + (long)bi * sCi;

    int mlo = lane & 15;
    int hi  = lane >> 4;
    int row0 = (tm << 5) + mlo;         // rows row0 and row0+16
    int n0   = tn << 5;

    const h16* Arow0 = Ab + (long)row0 * lda;
    const h16* Arow1 = Arow0 + (long)16 * lda;
    const h16* B0    = Bb + (long)(n0 + mlo)      * ldb;
    const h16* B1    = Bb + (long)(n0 + 16 + mlo) * ldb;
    int ka0 = hi ? 16 : 0;
    int ka1 = hi ? 24 : 8;
    int kb  = hi ? 16 : 0;

    v8f acc00 = {}, acc01 = {}, acc10 = {}, acc11 = {};
    for (int kk = 0; kk < Kr; kk += 32) {
        v8h a0lo = *(const v8h*)(Arow0 + kk + ka0);
        v8h a0hi = *(const v8h*)(Arow0 + kk + ka1);
        v16h av0 = __builtin_shufflevector(a0lo, a0hi, 0,1,2,3,4,5,6,7,8,9,10,11,12,13,14,15);
        v8h a1lo = *(const v8h*)(Arow1 + kk + ka0);
        v8h a1hi = *(const v8h*)(Arow1 + kk + ka1);
        v16h av1 = __builtin_shufflevector(a1lo, a1hi, 0,1,2,3,4,5,6,7,8,9,10,11,12,13,14,15);
        v16h b0 = *(const v16h*)(B0 + kk + kb);
        v16h b1 = *(const v16h*)(B1 + kk + kb);
        acc00 = __builtin_amdgcn_wmma_f32_16x16x32_f16(false, av0, false, b0, (short)0, acc00, false, false);
        acc01 = __builtin_amdgcn_wmma_f32_16x16x32_f16(false, av0, false, b1, (short)0, acc01, false, false);
        acc10 = __builtin_amdgcn_wmma_f32_16x16x32_f16(false, av1, false, b0, (short)0, acc10, false, false);
        acc11 = __builtin_amdgcn_wmma_f32_16x16x32_f16(false, av1, false, b1, (short)0, acc11, false, false);
    }

    int nc0 = n0 + mlo, nc1 = n0 + 16 + mlo;
    float bv0 = bias ? bias[nc0] : 0.f;
    float bv1 = bias ? bias[nc1] : 0.f;
    float* Cb = C + cOff;
    h16*   Db = C16 ? (C16 + cOff) : (h16*)0;

#pragma unroll
    for (int ms = 0; ms < 2; ++ms) {
        int mb = (tm << 5) + (ms << 4) + (hi << 3);
        v8f a0 = ms ? acc10 : acc00;
        v8f a1 = ms ? acc11 : acc01;
#pragma unroll
        for (int v = 0; v < 8; ++v) {
            long o0 = (long)(mb + v) * ldc + nc0;
            long o1 = (long)(mb + v) * ldc + nc1;
            float x0 = a0[v] + bv0;
            float x1 = a1[v] + bv1;
            if (accum) { x0 += Cb[o0]; x1 += Cb[o1]; }
            if (act == 1) { x0 = fmaxf(x0, 0.f); x1 = fmaxf(x1, 0.f); }
            else if (act == 2) { x0 = x0 > 0.f ? x0 : (__expf(x0) - 1.f);
                                 x1 = x1 > 0.f ? x1 : (__expf(x1) - 1.f); }
            Cb[o0] = x0; Cb[o1] = x1;
            if (Db) { Db[o0] = (h16)x0; Db[o1] = (h16)x1; }
        }
    }
}

// -------------------------- small helper kernels ---------------------------
__global__ void convert_w(const float* __restrict__ W, h16* __restrict__ Wt, int K, int N)
{
    int idx = blockIdx.x * 256 + threadIdx.x;
    if (idx >= K * N) return;
    int k = idx / N, n = idx - k * N;
    Wt[(long)n * K + k] = (h16)W[idx];
}

__global__ void embed_kernel(const float* __restrict__ x, const float* __restrict__ W,
                             const float* __restrict__ bias,
                             float* __restrict__ out, h16* __restrict__ out16,
                             int rows, int Din)
{
    int idx = blockIdx.x * 256 + threadIdx.x;
    if (idx >= rows * E_) return;
    int e = idx & (E_ - 1);
    int r = idx >> 8;
    float s = bias[e];
    for (int d = 0; d < Din; ++d) s = fmaf(x[r * Din + d], W[d * E_ + e], s);
    out[idx] = s; out16[idx] = (h16)s;
}

// (B,S,E) f32 -> (B,H,Dh,S) f16  (transposed per head, for WMMA B operand)
__global__ void transpose_heads(const float* __restrict__ v, h16* __restrict__ vt,
                                int S, int H, int Dh)
{
    int idx = blockIdx.x * 256 + threadIdx.x;
    if (idx >= B_ * S * E_) return;
    int e = idx & (E_ - 1);
    int s = (idx >> 8) % S;
    int b = idx / (S * E_);
    int h = e / Dh, d = e - h * Dh;
    vt[(((long)b * H + h) * Dh + d) * S + s] = (h16)v[idx];
}

__global__ void gat_src_dst(const float* __restrict__ h, const float* __restrict__ asrc,
                            const float* __restrict__ adst,
                            float* __restrict__ es, float* __restrict__ ed, int S)
{
    int idx = blockIdx.x * 256 + threadIdx.x;
    if (idx >= B_ * S * 4) return;
    int hh = idx & 3;
    int s  = (idx >> 2) % S;
    int b  = idx / (S * 4);
    const float* hp = h + ((long)(b * S + s)) * E_ + hh * 64;
    float s1 = 0.f, s2 = 0.f;
    for (int d = 0; d < 64; ++d) {
        float hv = hp[d];
        s1 = fmaf(hv, asrc[hh * 64 + d], s1);
        s2 = fmaf(hv, adst[hh * 64 + d], s2);
    }
    es[idx] = s1; ed[idx] = s2;
}

__global__ void gat_softmax(const float* __restrict__ es, const float* __restrict__ ed,
                            const unsigned char* __restrict__ mask,
                            h16* __restrict__ alpha, int S)
{
    __shared__ float sb[128];
    int blk = blockIdx.x;
    int i  = blk % S;
    int hh = (blk / S) & 3;
    int b  = blk / (S * 4);
    int j  = threadIdx.x;
    float e = es[((long)(b * S + i)) * 4 + hh] + ed[((long)(b * S + j)) * 4 + hh];
    e = e > 0.f ? e : 0.2f * e;                       // leaky_relu(0.2)
    if (mask && mask[((long)b * S + i) * S + j] == 0) e = -1e9f;
    sb[j] = e; __syncthreads();
    for (int st = S >> 1; st > 0; st >>= 1) { if (j < st) sb[j] = fmaxf(sb[j], sb[j + st]); __syncthreads(); }
    float mx = sb[0]; __syncthreads();
    float ex = __expf(e - mx);
    sb[j] = ex; __syncthreads();
    for (int st = S >> 1; st > 0; st >>= 1) { if (j < st) sb[j] += sb[j + st]; __syncthreads(); }
    alpha[(((long)(b * 4 + hh)) * S + i) * S + j] = (h16)(ex / sb[0]);
}

__global__ void xf_scores(const float* __restrict__ q, const float* __restrict__ k,
                          h16* __restrict__ att, int S, float scale)
{
    __shared__ float sb[128];
    int blk = blockIdx.x;
    int i  = blk % S;
    int hh = (blk / S) & 7;
    int b  = blk / (S * 8);
    int j  = threadIdx.x;
    const float* qp = q + ((long)(b * S + i)) * E_ + hh * 32;
    const float* kp = k + ((long)(b * S + j)) * E_ + hh * 32;
    float s = 0.f;
    for (int d = 0; d < 32; ++d) s = fmaf(qp[d], kp[d], s);
    s *= scale;
    sb[j] = s; __syncthreads();
    for (int st = S >> 1; st > 0; st >>= 1) { if (j < st) sb[j] = fmaxf(sb[j], sb[j + st]); __syncthreads(); }
    float mx = sb[0]; __syncthreads();
    float ex = __expf(s - mx);
    sb[j] = ex; __syncthreads();
    for (int st = S >> 1; st > 0; st >>= 1) { if (j < st) sb[j] += sb[j + st]; __syncthreads(); }
    att[(((long)(b * 8 + hh)) * S + i) * S + j] = (h16)(ex / sb[0]);
}

// out = LN(x + r) * g + beta, row length = 256
__global__ __launch_bounds__(256)
void ln_residual(const float* __restrict__ x, const float* __restrict__ r,
                 const float* __restrict__ g, const float* __restrict__ bta,
                 float* __restrict__ out, h16* __restrict__ out16)
{
    __shared__ float sb[256];
    int row = blockIdx.x, t = threadIdx.x;
    long off = (long)row * E_ + t;
    float v = x[off] + (r ? r[off] : 0.f);
    sb[t] = v; __syncthreads();
    for (int st = 128; st > 0; st >>= 1) { if (t < st) sb[t] += sb[t + st]; __syncthreads(); }
    float mu = sb[0] * (1.f / E_); __syncthreads();
    float d = v - mu;
    sb[t] = d * d; __syncthreads();
    for (int st = 128; st > 0; st >>= 1) { if (t < st) sb[t] += sb[t + st]; __syncthreads(); }
    float y = d * rsqrtf(sb[0] * (1.f / E_) + 1e-5f) * g[t] + bta[t];
    out[off] = y; out16[off] = (h16)y;
}

__global__ void dist_kernel(const float* __restrict__ rf, const float* __restrict__ tf,
                            float* __restrict__ dist, unsigned* __restrict__ maxbits)
{
    __shared__ float sb[256];
    int idx = blockIdx.x * 256 + threadIdx.x;   // exactly B*N*M
    int j = idx % M_;
    int i = (idx / M_) % N_;
    int b = idx / (M_ * N_);
    float dx = rf[((b * N_) + i) * 6 + 0] - tf[((b * M_) + j) * 9 + 0];
    float dy = rf[((b * N_) + i) * 6 + 1] - tf[((b * M_) + j) * 9 + 1];
    float d = sqrtf(dx * dx + dy * dy);
    dist[idx] = d;
    sb[threadIdx.x] = d; __syncthreads();
    for (int st = 128; st > 0; st >>= 1) {
        if (threadIdx.x < st) sb[threadIdx.x] = fmaxf(sb[threadIdx.x], sb[threadIdx.x + st]);
        __syncthreads();
    }
    if (threadIdx.x == 0) atomicMax(maxbits, __float_as_uint(sb[0]));
}

// Fused pair epilogue: reward[b,i,j] and idle-sum[b,i].
__global__ __launch_bounds__(256)
void final_kernel(const float* __restrict__ ArR, const float* __restrict__ AtR,
                  const float* __restrict__ ArI, const float* __restrict__ AtI,
                  const float* __restrict__ b1R, const float* __restrict__ W2R,
                  const float* __restrict__ b2R, const float* __restrict__ wlR,
                  const float* __restrict__ b1I, const float* __restrict__ W2I,
                  const float* __restrict__ b2I, const float* __restrict__ wlI,
                  const float* __restrict__ dist, const unsigned* __restrict__ maxbits,
                  const float* __restrict__ dW1, const float* __restrict__ db1,
                  const float* __restrict__ dW2, const float* __restrict__ db2,
                  float* __restrict__ out)
{
    __shared__ float redR[8], redI[8];
    int blk = blockIdx.x;            // b*64 + i
    int b = blk >> 6;
    int t = threadIdx.x;
    int wid = t >> 5;

    float ar0 = ArR[(long)blk * FF_ + t],       ar1 = ArR[(long)blk * FF_ + 256 + t];
    float ai0 = ArI[(long)blk * FF_ + t],       ai1 = ArI[(long)blk * FF_ + 256 + t];
    float bR0 = b1R[t], bR1 = b1R[256 + t];
    float bI0 = b1I[t], bI1 = b1I[256 + t];
    float wR0 = W2R[t], wR1 = W2R[256 + t];
    float wI0 = W2I[t], wI1 = W2I[256 + t];
    float lR0 = wlR[t], lR1 = wlR[256 + t];
    float lI0 = wlI[t], lI1 = wlI[256 + t];
    float invmax = 1.f / __uint_as_float(*maxbits);
    float mW1[16], mb1[16], mW2[16];
#pragma unroll
    for (int h = 0; h < 16; ++h) { mW1[h] = dW1[h]; mb1[h] = db1[h]; mW2[h] = dW2[h]; }
    float db2v = db2[0], b2Rv = b2R[0], b2Iv = b2I[0];
    float idle = 0.f;

    for (int j = 0; j < M_; ++j) {
        if (j + 1 < M_)
            __builtin_prefetch(&AtR[((long)b * M_ + j + 1) * FF_ + t], 0, 3); // global_prefetch_b8
        float rel = dist[(long)blk * M_ + j] * invmax;
        float pd = db2v;
#pragma unroll
        for (int h = 0; h < 16; ++h) {
            float u = fmaf(rel, mW1[h], mb1[h]);
            pd = fmaf(fmaxf(u, 0.f), mW2[h], pd);
        }
        long ao = ((long)b * M_ + j) * FF_;
        float at0 = AtR[ao + t], at1 = AtR[ao + 256 + t];
        float bt0 = AtI[ao + t], bt1 = AtI[ao + 256 + t];
        float h0 = fmaxf(ar0 + at0 + pd * lR0 + bR0, 0.f);
        float h1 = fmaxf(ar1 + at1 + pd * lR1 + bR1, 0.f);
        float sr = h0 * wR0 + h1 * wR1;
        float g0 = fmaxf(ai0 + bt0 + pd * lI0 + bI0, 0.f);
        float g1 = fmaxf(ai1 + bt1 + pd * lI1 + bI1, 0.f);
        float si = g0 * wI0 + g1 * wI1;
        for (int o = 16; o > 0; o >>= 1) { sr += __shfl_down(sr, o); si += __shfl_down(si, o); }
        if ((t & 31) == 0) { redR[wid] = sr; redI[wid] = si; }
        __syncthreads();
        if (t == 0) {
            float R = 0.f, I = 0.f;
#pragma unroll
            for (int w = 0; w < 8; ++w) { R += redR[w]; I += redI[w]; }
            out[(long)blk * (M_ + 1) + j] = R + b2Rv;
            idle += I + b2Iv;
        }
        __syncthreads();
    }
    if (t == 0) out[(long)blk * (M_ + 1) + M_] = idle;
}

// ------------------------------- host side ---------------------------------
struct GatP { const float *W, *adst, *asrc; };
struct XfP  { const float *W1,*W2,*Wk,*Wo,*Wq,*Wv,*b1,*b2,*be1,*be2,*bk,*bo,*bq,*bv,*g1,*g2; };
struct XfW  { h16 *Wq,*Wk,*Wv,*Wo,*W1,*W2; };

extern "C" void kernel_launch(void* const* d_in, const int* in_sizes, int n_in,
                              void* d_out, int out_size, void* d_ws, size_t ws_size,
                              hipStream_t stream)
{
    (void)in_sizes; (void)out_size; (void)ws_size;
    if (n_in < 95) return;
    auto F = [&](int i) { return (const float*)d_in[i]; };

    // jax pytree flatten order (dict keys sorted, lists in order):
    const float* rf  = F(0);                       // robot_features (8,64,6)
    const float* tf  = F(1);                       // task_features  (8,128,9)
    const unsigned char* adj = (const unsigned char*)d_in[2]; // (8,128,128) bool
    const float *dW1 = F(3),  *dW2 = F(4),  *db1 = F(5),  *db2 = F(6);   // dist_mlp
    const float *iW1 = F(7),  *iW2 = F(8),  *ib1 = F(9),  *ib2 = F(10);  // idle_mlp
    const float *rW1 = F(11), *rW2 = F(12), *rb1 = F(13), *rb2 = F(14);  // reward_mlp
    const float *reW = F(15), *reB = F(16);                              // robot_emb
    GatP rgat[2] = {{F(17),F(18),F(19)},{F(20),F(21),F(22)}};
    XfP  rxf[2], txf[2];
    for (int l = 0; l < 2; ++l) { const float** p = (const float**)&rxf[l]; for (int q = 0; q < 16; ++q) p[q] = F(23 + 16*l + q); }
    const float *teW = F(55), *teB = F(56);                              // task_emb
    GatP tgat[2] = {{F(57),F(58),F(59)},{F(60),F(61),F(62)}};
    for (int l = 0; l < 2; ++l) { const float** p = (const float**)&txf[l]; for (int q = 0; q < 16; ++q) p[q] = F(63 + 16*l + q); }

    // bump allocator over d_ws
    char* wsb = (char*)d_ws; size_t off = 0;
    auto alloc = [&](size_t nbytes) -> void* {
        void* p = wsb + off; off = (off + nbytes + 255) & ~(size_t)255; return p; };
    auto fbuf = [&](size_t n) -> float* { return (float*)alloc(n * 4); };
    auto hbuf = [&](size_t n) -> h16*   { return (h16*)alloc(n * 2); };

    auto convw = [&](const float* W, int K, int N) -> h16* {
        h16* Wt = hbuf((size_t)K * N);
        int tot = K * N;
        convert_w<<<(tot + 255) / 256, 256, 0, stream>>>(W, Wt, K, N);
        return Wt;
    };

    auto gemm = [&](const h16* A, int lda, long sAo, long sAi,
                    const h16* Bt, int ldb, long sBo, long sBi,
                    float* C, int ldc, long sCo, long sCi, h16* C16,
                    const float* bias, int Mr, int Nr, int Kr,
                    int BO, int BI, int act, int acc) {
        int waves = BO * BI * (Mr / 32) * (Nr / 32);
        wmma_gemm<<<(waves + 7) / 8, 256, 0, stream>>>(
            A, lda, sAo, sAi, Bt, ldb, sBo, sBi, C, ldc, sCo, sCi,
            C16, bias, Mr, Nr, Kr, BO, BI, act, acc);
    };

    // ---- weight conversion (f32 -> transposed f16) ----
    h16* rgatWt[2] = { convw(rgat[0].W, 256, 256), convw(rgat[1].W, 256, 256) };
    h16* tgatWt[2] = { convw(tgat[0].W, 256, 256), convw(tgat[1].W, 256, 256) };
    XfW rxfW[2], txfW[2];
    auto buildXfW = [&](XfP* xp, XfW* o) {
        for (int l = 0; l < 2; ++l) {
            o[l].Wq = convw(xp[l].Wq, 256, 256); o[l].Wk = convw(xp[l].Wk, 256, 256);
            o[l].Wv = convw(xp[l].Wv, 256, 256); o[l].Wo = convw(xp[l].Wo, 256, 256);
            o[l].W1 = convw(xp[l].W1, 256, 512); o[l].W2 = convw(xp[l].W2, 512, 256);
        }
    };
    buildXfW(rxf, rxfW); buildXfW(txf, txfW);

    // ---- per-stream pipeline: embed -> GAT x2 -> XF x2 ----
    auto run_stream = [&](int S, int Din, const float* feat,
                          const float* embW, const float* embB,
                          GatP* gat, h16** gatWt, XfP* xf, XfW* xfw,
                          const unsigned char* mask, h16** rg16_out, h16** ro16_out) {
        int R = B_ * S, tot = R * E_;
        float* xA = fbuf((size_t)tot); h16* xA16 = hbuf((size_t)tot);
        float* xB = fbuf((size_t)tot); h16* xB16 = hbuf((size_t)tot);
        float* xC = fbuf((size_t)tot); h16* xC16 = hbuf((size_t)tot);
        float* xD = fbuf((size_t)tot); h16* xD16 = hbuf((size_t)tot);
        float* hB = fbuf((size_t)tot);
        float* qB = fbuf((size_t)tot); float* kB = fbuf((size_t)tot);
        float* vB = fbuf((size_t)tot); float* oB = fbuf((size_t)tot);
        float* t1 = fbuf((size_t)tot);
        h16* o16  = hbuf((size_t)tot);
        h16* ht16 = hbuf((size_t)tot);
        h16* vt16 = hbuf((size_t)tot);
        float* f1 = fbuf((size_t)R * FF_); h16* f116 = hbuf((size_t)R * FF_);
        h16* alpha = hbuf((size_t)B_ * 4 * S * S);
        h16* att   = hbuf((size_t)B_ * 8 * S * S);
        float* es = fbuf((size_t)R * 4); float* ed = fbuf((size_t)R * 4);

        embed_kernel<<<(tot + 255) / 256, 256, 0, stream>>>(feat, embW, embB, xA, xA16, R, Din);

        float* cur = xA; h16* cur16 = xA16;
        float* nxt = xB; h16* nxt16 = xB16;
        for (int l = 0; l < 2; ++l) {
            gemm(cur16, 256, 0, 0, gatWt[l], 256, 0, 0,
                 hB, 256, 0, 0, nullptr, nullptr, R, 256, 256, 1, 1, 0, 0);
            transpose_heads<<<(tot + 255) / 256, 256, 0, stream>>>(hB, ht16, S, 4, 64);
            gat_src_dst<<<(R * 4 + 255) / 256, 256, 0, stream>>>(hB, gat[l].asrc, gat[l].adst, es, ed, S);
            gat_softmax<<<B_ * 4 * S, S, 0, stream>>>(es, ed, mask, alpha, S);
            gemm(alpha, S, (long)4 * S * S, (long)S * S,
                 ht16,  S, (long)4 * 64 * S, (long)64 * S,
                 nxt, 256, (long)S * 256, 64, nxt16, nullptr,
                 S, 64, S, B_, 4, /*elu*/2, 0);
            float* tf_ = cur; cur = nxt; nxt = tf_;
            h16* th_ = cur16; cur16 = nxt16; nxt16 = th_;
        }
        *rg16_out = cur16;   // GAT output ends in xA (2 layers)

        for (int l = 0; l < 2; ++l) {
            XfW& w = xfw[l]; XfP& p = xf[l];
            gemm(cur16, 256, 0, 0, w.Wq, 256, 0, 0, qB, 256, 0, 0, nullptr, p.bq, R, 256, 256, 1, 1, 0, 0);
            gemm(cur16, 256, 0, 0, w.Wk, 256, 0, 0, kB, 256, 0, 0, nullptr, p.bk, R, 256, 256, 1, 1, 0, 0);
            gemm(cur16, 256, 0, 0, w.Wv, 256, 0, 0, vB, 256, 0, 0, nullptr, p.bv, R, 256, 256, 1, 1, 0, 0);
            transpose_heads<<<(tot + 255) / 256, 256, 0, stream>>>(vB, vt16, S, 8, 32);
            xf_scores<<<B_ * 8 * S, S, 0, stream>>>(qB, kB, att, S, 0.1767766953f); // 1/sqrt(32)
            gemm(att, S, (long)8 * S * S, (long)S * S,
                 vt16, S, (long)256 * S, (long)32 * S,
                 oB, 256, (long)S * 256, 32, o16, nullptr,
                 S, 32, S, B_, 8, 0, 0);
            gemm(o16, 256, 0, 0, w.Wo, 256, 0, 0, t1, 256, 0, 0, nullptr, p.bo, R, 256, 256, 1, 1, 0, 0);
            ln_residual<<<R, 256, 0, stream>>>(t1, cur, p.g1, p.be1, xC, xC16);
            gemm(xC16, 256, 0, 0, w.W1, 256, 0, 0, f1, 512, 0, 0, f116, p.b1, R, 512, 256, 1, 1, /*relu*/1, 0);
            gemm(f116, 512, 0, 0, w.W2, 512, 0, 0, t1, 256, 0, 0, nullptr, p.b2, R, 256, 512, 1, 1, 0, 0);
            float* dst = (l == 0) ? xD : xB;
            h16*  dst16 = (l == 0) ? xD16 : xB16;
            ln_residual<<<R, 256, 0, stream>>>(t1, xC, p.g2, p.be2, dst, dst16);
            cur = dst; cur16 = dst16;
        }
        *ro16_out = cur16;
    };

    h16 *rg16, *ro16, *tg16, *to16;
    run_stream(N_, 6, rf, reW, reB, rgat, rgatWt, rxf, rxfW, nullptr, &rg16, &ro16);
    run_stream(M_, 9, tf, teW, teB, tgat, tgatWt, txf, txfW, adj,     &tg16, &to16);

    // ---- factorized reward/idle W1 blocks (each 256x512) ----
    h16* rwA = convw(rW1 +   0 * 512, 256, 512);
    h16* rwB = convw(rW1 + 256 * 512, 256, 512);
    h16* rwC = convw(rW1 + 512 * 512, 256, 512);
    h16* rwD = convw(rW1 + 768 * 512, 256, 512);
    h16* iwA = convw(iW1 +   0 * 512, 256, 512);
    h16* iwB = convw(iW1 + 256 * 512, 256, 512);
    h16* iwC = convw(iW1 + 512 * 512, 256, 512);
    h16* iwD = convw(iW1 + 768 * 512, 256, 512);

    float* ArR = fbuf((size_t)B_ * N_ * FF_);
    float* AtR = fbuf((size_t)B_ * M_ * FF_);
    float* ArI = fbuf((size_t)B_ * N_ * FF_);
    float* AtI = fbuf((size_t)B_ * M_ * FF_);
    gemm(rg16, 256, 0, 0, rwA, 256, 0, 0, ArR, 512, 0, 0, nullptr, nullptr, B_*N_, 512, 256, 1, 1, 0, 0);
    gemm(ro16, 256, 0, 0, rwC, 256, 0, 0, ArR, 512, 0, 0, nullptr, nullptr, B_*N_, 512, 256, 1, 1, 0, 1);
    gemm(tg16, 256, 0, 0, rwB, 256, 0, 0, AtR, 512, 0, 0, nullptr, nullptr, B_*M_, 512, 256, 1, 1, 0, 0);
    gemm(to16, 256, 0, 0, rwD, 256, 0, 0, AtR, 512, 0, 0, nullptr, nullptr, B_*M_, 512, 256, 1, 1, 0, 1);
    gemm(rg16, 256, 0, 0, iwA, 256, 0, 0, ArI, 512, 0, 0, nullptr, nullptr, B_*N_, 512, 256, 1, 1, 0, 0);
    gemm(ro16, 256, 0, 0, iwC, 256, 0, 0, ArI, 512, 0, 0, nullptr, nullptr, B_*N_, 512, 256, 1, 1, 0, 1);
    gemm(tg16, 256, 0, 0, iwB, 256, 0, 0, AtI, 512, 0, 0, nullptr, nullptr, B_*M_, 512, 256, 1, 1, 0, 0);
    gemm(to16, 256, 0, 0, iwD, 256, 0, 0, AtI, 512, 0, 0, nullptr, nullptr, B_*M_, 512, 256, 1, 1, 0, 1);

    // ---- distances + global max ----
    float* dist = fbuf((size_t)B_ * N_ * M_);
    unsigned* maxbits = (unsigned*)alloc(256);
    hipMemsetAsync(maxbits, 0, sizeof(unsigned), stream);
    dist_kernel<<<(B_ * N_ * M_) / 256, 256, 0, stream>>>(rf, tf, dist, maxbits);

    // ---- fused pair epilogue ----
    final_kernel<<<B_ * N_, 256, 0, stream>>>(
        ArR, AtR, ArI, AtI,
        rb1, rW2, rb2, rW1 + (size_t)1024 * 512,
        ib1, iW2, ib2, iW1 + (size_t)1024 * 512,
        dist, maxbits, dW1, db1, dW2, db2, (float*)d_out);
}
